// NearestPrior_49538152792860
// MI455X (gfx1250) — matrix operand order repaired and checked by reading, not compiled, not run on hardware
//
#include <hip/hip_runtime.h>
#include <hip/hip_bf16.h>
#include <math.h>

// Problem constants (from reference setup_inputs)
#define N_TOT  8192
#define D_FEAT 256
#define N_CLS  65
#define S_SRC  4096
#define K_NN   5

typedef __attribute__((ext_vector_type(16))) __bf16 v16bf;
typedef __attribute__((ext_vector_type(8)))  __bf16 v8bf;
typedef __attribute__((ext_vector_type(8)))  float  v8f;

__device__ __forceinline__ unsigned short f32_to_bf16(float f) {
    unsigned u = __builtin_bit_cast(unsigned, f);
    unsigned r = u + 0x7fffu + ((u >> 16) & 1u);   // round-to-nearest-even
    return (unsigned short)(r >> 16);
}
__device__ __forceinline__ float bf16_to_f32(unsigned short h) {
    unsigned u = ((unsigned)h) << 16;
    return __builtin_bit_cast(float, u);
}
__device__ __forceinline__ v8bf ld8(const unsigned short* p) {
    return *(const v8bf*)p;
}

// Branchless insert of t into ascending 5-list (c0..c4 = 5 smallest seen).
__device__ __forceinline__ void insert5(float& c0, float& c1, float& c2,
                                        float& c3, float& c4, float t) {
    float t4 = fminf(c4, t);                       // candidate for slot 4 (evict old max)
    float a  = fminf(c3, t4);  c4 = fmaxf(c3, t4); // bubble down
    float b  = fminf(c2, a);   c3 = fmaxf(c2, a);
    float d  = fminf(c1, b);   c2 = fmaxf(c1, b);
    float e  = fminf(c0, d);   c1 = fmaxf(c0, d);
    c0 = e;
}

// ---------------------------------------------------------------- init
__global__ void k_init(float* out) {
    if (threadIdx.x < 3) out[threadIdx.x] = 0.0f;
}

// ------------------------------------------- prep: bf16 hi/lo split + row norms
__global__ void k_prep(const float* __restrict__ X,
                       unsigned short* __restrict__ Xhi,
                       unsigned short* __restrict__ Xlo,
                       float* __restrict__ sq) {
    __shared__ float red[D_FEAT];
    const int r = blockIdx.x, t = threadIdx.x;
    float f = X[(size_t)r * D_FEAT + t];
    unsigned short h = f32_to_bf16(f);
    unsigned short l = f32_to_bf16(f - bf16_to_f32(h));
    Xhi[(size_t)r * D_FEAT + t] = h;
    Xlo[(size_t)r * D_FEAT + t] = l;
    red[t] = f * f;
    __syncthreads();
    for (int s = D_FEAT / 2; s > 0; s >>= 1) {
        if (t < s) red[t] += red[t + s];
        __syncthreads();
    }
    if (t == 0) sq[r] = red[0];
}

// ------------------------------------------- loss_cl + loss_ce (row softmax reductions)
__global__ void k_logit_losses(const float* __restrict__ logit,
                               const int* __restrict__ y,
                               float* __restrict__ out) {
    const int row = blockIdx.x * blockDim.x + threadIdx.x;
    if (row >= N_TOT) return;
    const float* p = logit + (size_t)row * N_CLS;
    float m = -__builtin_inff();
    for (int j = 0; j < N_CLS; ++j) m = fmaxf(m, p[j]);
    float s = 0.0f;
    for (int j = 0; j < N_CLS; ++j) s += expf(p[j] - m);
    const float lse = m + logf(s);
    if (row < S_SRC) {
        // -log_softmax picked at label, natural log
        atomicAdd(&out[0], (lse - p[y[row]]) * (1.0f / (float)S_SRC));
    } else {
        // entropy in bits of softmax row, p clipped at 1e-8
        float H = 0.0f;
        for (int j = 0; j < N_CLS; ++j) {
            float pj = expf(p[j] - lse);
            H -= pj * log2f(fmaxf(pj, 1e-8f));
        }
        atomicAdd(&out[1], H * (1.0f / (float)(N_TOT - S_SRC)));
    }
}

// ------------------------------------------- fused Gram-GEMM + streaming 5-NN + loss_reg
// grid = N_TOT/16 blocks of 256 threads (8 waves). Block owns 16 rows; each wave
// streams 32 of the 256 column tiles per half. bf16x3 split recovers ~fp32 accuracy.
__global__ void __launch_bounds__(256)
k_knn_reg(const unsigned short* __restrict__ Xhi,
          const unsigned short* __restrict__ Xlo,
          const float* __restrict__ sq,
          float* __restrict__ out) {
    __shared__ float smTop[16][128][K_NN];   // 40 KB: 128 partial 5-lists per row
    __shared__ float smD[2][16][3];          // per half, per row: d0, d1, d4

    const int tid   = threadIdx.x;
    const int lane  = tid & 31;
    const int wave  = tid >> 5;
    const int rowBase = blockIdx.x * 16;
    const int mn    = lane & 15;     // m for A rows, n for B/C columns
    const int khalf = lane >> 4;     // K half-select for A/B 16-bit layouts

    // ---- A tile (16 rows x 256 K) resident in VGPRs, hi + lo planes.
    // 16-bit A layout: per lane, elems 0..7 = K kb+khalf*8 .. +7, elems 8..15 = K kb+16+khalf*8 .. +7
    v16bf ah[8], al[8];
    {
        const unsigned short* rh = Xhi + (size_t)(rowBase + mn) * D_FEAT;
        const unsigned short* rl = Xlo + (size_t)(rowBase + mn) * D_FEAT;
        #pragma unroll
        for (int q = 0; q < 8; ++q) {
            const int kb = q * 32 + khalf * 8;
            v8bf h0 = ld8(rh + kb), h1 = ld8(rh + kb + 16);
            v8bf l0 = ld8(rl + kb), l1 = ld8(rl + kb + 16);
            #pragma unroll
            for (int e = 0; e < 8; ++e) {
                ah[q][e] = h0[e]; ah[q][e + 8] = h1[e];
                al[q][e] = l0[e]; al[q][e + 8] = l1[e];
            }
        }
    }

    for (int h = 0; h < 2; ++h) {                 // column half: [0,S) then [S,N)
        float b0[8], b1[8], b2[8], b3[8], b4[8];  // per-lane top-5 of t for its 8 rows
        #pragma unroll
        for (int v = 0; v < 8; ++v)
            b0[v] = b1[v] = b2[v] = b3[v] = b4[v] = __builtin_inff();

        for (int it = 0; it < 32; ++it) {
            const int colBase = h * S_SRC + (it * 8 + wave) * 16;
            const int col = colBase + mn;
            const unsigned short* ch = Xhi + (size_t)col * D_FEAT;
            const unsigned short* cl = Xlo + (size_t)col * D_FEAT;
            v8f c = {};
            #pragma unroll
            for (int q = 0; q < 8; ++q) {
                // 16-bit B layout: per lane, elems 0..7 = K kb .. +7, elems 8..15 = K kb+16 .. +7, kb = q*32 + khalf*8
                const int kb = q * 32 + khalf * 8;
                v8bf h0 = ld8(ch + kb), h1 = ld8(ch + kb + 16);
                v8bf l0 = ld8(cl + kb), l1 = ld8(cl + kb + 16);
                v16bf bh, bl;
                #pragma unroll
                for (int e = 0; e < 8; ++e) {
                    bh[e] = h0[e]; bh[e + 8] = h1[e];
                    bl[e] = l0[e]; bl[e + 8] = l1[e];
                }
                // dot = hi*hi + hi*lo + lo*hi  (bf16x3 fp32 emulation), f32 accumulate
                c = __builtin_amdgcn_wmma_f32_16x16x32_bf16(false, ah[q], false, bh, (short)0, c, false, false);
                c = __builtin_amdgcn_wmma_f32_16x16x32_bf16(false, ah[q], false, bl, (short)0, c, false, false);
                c = __builtin_amdgcn_wmma_f32_16x16x32_bf16(false, al[q], false, bh, (short)0, c, false, false);
            }
            const float sqc = sq[col];
            #pragma unroll
            for (int v = 0; v < 8; ++v) {
                // t = |x_j|^2 - 2 x_i.x_j  (row-constant |x_i|^2 dropped: order-preserving)
                const float t = fmaf(-2.0f, c[v], sqc);
                insert5(b0[v], b1[v], b2[v], b3[v], b4[v], t);
            }
        }

        // ---- dump partial lists: row m = v + 8*(lane>>4), list id = wave*16 + n
        {
            const int mh = lane >> 4;
            #pragma unroll
            for (int v = 0; v < 8; ++v) {
                const int m  = v + 8 * mh;
                const int li = wave * 16 + mn;
                smTop[m][li][0] = b0[v]; smTop[m][li][1] = b1[v];
                smTop[m][li][2] = b2[v]; smTop[m][li][3] = b3[v];
                smTop[m][li][4] = b4[v];
            }
        }
        __syncthreads();
        if (tid < 16) {
            float c0, c1, c2, c3, c4;
            c0 = c1 = c2 = c3 = c4 = __builtin_inff();
            for (int li = 0; li < 128; ++li) {
                #pragma unroll
                for (int j = 0; j < K_NN; ++j)
                    insert5(c0, c1, c2, c3, c4, smTop[tid][li][j]);
            }
            const float srow = sq[rowBase + tid];
            // d_k = sqrt(max(|x_i|^2 + t_k, 0))
            smD[h][tid][0] = sqrtf(fmaxf(srow + c0, 0.0f));
            smD[h][tid][1] = sqrtf(fmaxf(srow + c1, 0.0f));
            smD[h][tid][2] = sqrtf(fmaxf(srow + c4, 0.0f));
        }
        __syncthreads();  // also protects smTop reuse in next half
    }

    if (tid < 16) {
        const int grow = rowBase + tid;
        const int hIntra = (grow < S_SRC) ? 0 : 1;
        const int hInter = 1 - hIntra;
        const float d1i = smD[hIntra][tid][1];  // min off-diag dist (diag is row min)
        const float d4i = smD[hIntra][tid][2];  // 5th smallest incl diag
        const float d0e = smD[hInter][tid][0];  // min dist (no diag in inter block)
        const float d4e = smD[hInter][tid][2];
        // sigma=1 -> denom = 2*var^2 = 2*(kth + 1e-8); max sim = exp(-d_min/denom)
        const float sIntra = expf(-d1i / (2.0f * (d4i + 1e-8f)));
        const float sInter = expf(-d0e / (2.0f * (d4e + 1e-8f)));
        atomicAdd(&out[2], fabsf(sIntra - sInter) * (1.0f / (float)S_SRC));
    }
}

// ---------------------------------------------------------------- launch
extern "C" void kernel_launch(void* const* d_in, const int* in_sizes, int n_in,
                              void* d_out, int out_size, void* d_ws, size_t ws_size,
                              hipStream_t stream) {
    (void)in_sizes; (void)n_in; (void)out_size; (void)ws_size;
    const float* X     = (const float*)d_in[0];  // 8192 x 256 fp32
    const float* logit = (const float*)d_in[1];  // 8192 x 65  fp32
    const int*   y     = (const int*)d_in[2];    // 4096 int32
    float* out = (float*)d_out;                  // {loss_cl, loss_ce, loss_reg}

    // ws layout: Xhi(4MB) | Xlo(4MB) | sq(32KB)  -> ~8.03 MB total
    unsigned short* Xhi = (unsigned short*)d_ws;
    unsigned short* Xlo = Xhi + (size_t)N_TOT * D_FEAT;
    float* sq = (float*)(Xlo + (size_t)N_TOT * D_FEAT);

    k_init<<<1, 32, 0, stream>>>(out);
    k_prep<<<N_TOT, D_FEAT, 0, stream>>>(X, Xhi, Xlo, sq);
    k_logit_losses<<<N_TOT / 256, 256, 0, stream>>>(logit, y, out);
    k_knn_reg<<<N_TOT / 16, 256, 0, stream>>>(Xhi, Xlo, sq, out);
}
